// MambaIntegerBlockV2Redesign_77472620085533
// MI455X (gfx1250) — matrix operand order repaired and checked by reading, not compiled, not run on hardware
//
#include <hip/hip_runtime.h>

typedef __attribute__((ext_vector_type(16))) __bf16 v16bf;
typedef __attribute__((ext_vector_type(8)))  float  v8f;
typedef __attribute__((ext_vector_type(8)))  __bf16 bf16x8;
typedef __attribute__((ext_vector_type(4)))  __bf16 bf16x4;

// Problem constants (from reference): B=16, H=32, CS=256, DH=64, DS=128
constexpr int kCS = 256;
constexpr int kDH = 64;
constexpr int kDS = 128;
constexpr float kALPHA = 0.5f;

// LDS strides (elements). Padded +8 bf16 (16B) -> rows stay 16B aligned and
// rotate 4 banks per row => conflict-free ds_load_b128 fragment loads.
constexpr int C_STR  = kDS + 8;   // 136
constexpr int B_STR  = kDS + 8;   // 136
constexpr int BT_STR = kCS + 8;   // 264
constexpr int XT_STR = kCS + 8;   // 264
constexpr int HT_STR = kDS + 8;   // 136
constexpr int SC_STR = 40;        // 32 + 8 (per-wave 16x32 scratch tile)

// Dynamic LDS layout (byte offsets, all 16B aligned)
constexpr size_t OFF_ACUM = 0;                                       // f32[256]
constexpr size_t OFF_W    = 1024;                                    // f32[256] exp(atot - acum)
constexpr size_t OFF_DF   = 2048;                                    // f32[256] exp(acum)
constexpr size_t OFF_CS   = 3072;
constexpr size_t OFF_BS   = OFF_CS  + size_t(kCS) * C_STR  * 2;
constexpr size_t OFF_BTW  = OFF_BS  + size_t(kCS) * B_STR  * 2;
constexpr size_t OFF_XT   = OFF_BTW + size_t(kDS) * BT_STR * 2;
constexpr size_t OFF_HT   = OFF_XT  + size_t(kDH) * XT_STR * 2;
constexpr size_t OFF_SCR  = OFF_HT  + size_t(kDH) * HT_STR * 2;
constexpr size_t SMEM_BYTES = OFF_SCR + size_t(8) * 16 * SC_STR * 2; // ~271 KB < 320 KB

// ---- Fragment loaders (CDNA5 wave32 layouts, CDNA5 ISA §7.12.2) ----
// A 16x32 bf16: lane L holds row M = m0 + (L&15); elems 0..7 -> K = k0+kb..+7,
// elems 8..15 -> K = k0+kb+16..+23, kb = (L>=16 ? 8 : 0). Two ds_load_b128.
__device__ __forceinline__ v16bf load_a16(const __bf16* m, int stride, int m0,
                                          int k0, int lane) {
  const int row = m0 + (lane & 15);
  const int kb  = k0 + ((lane & 16) ? 8 : 0);
  const __bf16* p = m + row * stride + kb;
  bf16x8 lo = *(const bf16x8*)(p);
  bf16x8 hi = *(const bf16x8*)(p + 16);
  return __builtin_shufflevector(lo, hi, 0, 1, 2, 3, 4, 5, 6, 7,
                                 8, 9, 10, 11, 12, 13, 14, 15);
}

// B 32x16 bf16: lane L holds col N = n0 + (L&15); elems 0..7 -> K = k0+kb..+7,
// elems 8..15 -> K = k0+kb+16..+23, kb = (L>=16 ? 8 : 0). Memory passed as m[N*stride + K]. Two ds_load_b128.
__device__ __forceinline__ v16bf load_b16(const __bf16* m, int stride, int n0,
                                          int k0, int lane) {
  const int col = n0 + (lane & 15);
  const int kb  = k0 + ((lane & 16) ? 8 : 0);
  const __bf16* p = m + col * stride + kb;
  bf16x8 lo = *(const bf16x8*)(p);
  bf16x8 hi = *(const bf16x8*)(p + 16);
  return __builtin_shufflevector(lo, hi, 0, 1, 2, 3, 4, 5, 6, 7,
                                 8, 9, 10, 11, 12, 13, 14, 15);
}

__device__ __forceinline__ v8f wmma_bf16(v16bf a, v16bf b, v8f c) {
  return __builtin_amdgcn_wmma_f32_16x16x32_bf16(false, a, false, b, (short)0,
                                                 c, false, false);
}

__global__ void __launch_bounds__(256, 1)
ssd_chunk_kernel(const float* __restrict__ Xg, const float* __restrict__ Ag,
                 const float* __restrict__ Bg, const float* __restrict__ Cg,
                 const float* __restrict__ Hg, float* __restrict__ Yout,
                 float* __restrict__ Hout, float* __restrict__ Dout) {
  extern __shared__ char smem[];
  float*  smAcum = (float*)(smem + OFF_ACUM);
  float*  smW    = (float*)(smem + OFF_W);
  float*  smDF   = (float*)(smem + OFF_DF);
  __bf16* smCs   = (__bf16*)(smem + OFF_CS);
  __bf16* smBs   = (__bf16*)(smem + OFF_BS);
  __bf16* smBTw  = (__bf16*)(smem + OFF_BTW);
  __bf16* smXT   = (__bf16*)(smem + OFF_XT);
  __bf16* smHT   = (__bf16*)(smem + OFF_HT);
  __bf16* smScr  = (__bf16*)(smem + OFF_SCR);

  const int tid  = threadIdx.x;
  const int lane = tid & 31;
  // Force wave index scalar: removes divergent-loop exec bookkeeping below and
  // guarantees EXEC==all-1s at every WMMA (ISA requirement).
  const int wave = __builtin_amdgcn_readfirstlane(tid >> 5);
  const int bh   = blockIdx.x;

  const float* X = Xg + (size_t)bh * kCS * kDH;
  const float* A = Ag + (size_t)bh * kCS;
  const float* B = Bg + (size_t)bh * kCS * kDS;
  const float* C = Cg + (size_t)bh * kCS * kDS;
  const float* H = Hg + (size_t)bh * kDS * kDH;

  // ---- Phase 1: inclusive cumsum of A over CS (Hillis-Steele in LDS) ----
  float av = A[tid];
  smAcum[tid] = av;
  __syncthreads();
#pragma unroll
  for (int off = 1; off < kCS; off <<= 1) {
    float add = (tid >= off) ? smAcum[tid - off] : 0.0f;
    __syncthreads();
    av += add;
    smAcum[tid] = av;
    __syncthreads();
  }
  const float atot = smAcum[kCS - 1];
  smW[tid]  = __expf(atot - av);  // decay_to_end[t]
  smDF[tid] = __expf(av);         // decay_from_start[t]
  __syncthreads();

  // ---- Phase 2: stage operands into LDS as bf16 (b128 global loads) ----
  {
    const float4* C4 = (const float4*)C;
    for (int k = tid; k < (kCS * kDS) / 4; k += 256) {
      const float4 v = C4[k];
      const int i = k >> 5, d = (k & 31) * 4;  // 32 float4 per 128-row
      bf16x4 p = {(__bf16)v.x, (__bf16)v.y, (__bf16)v.z, (__bf16)v.w};
      *(bf16x4*)(&smCs[i * C_STR + d]) = p;    // ds_store_b64
    }
    const float4* B4 = (const float4*)B;
    for (int k = tid; k < (kCS * kDS) / 4; k += 256) {
      const float4 v = B4[k];
      const int j = k >> 5, s = (k & 31) * 4;
      bf16x4 p = {(__bf16)v.x, (__bf16)v.y, (__bf16)v.z, (__bf16)v.w};
      *(bf16x4*)(&smBs[j * B_STR + s]) = p;    // for CB = C.B^T
      const float wj = smW[j];                 // B^T with decay folded
      smBTw[(s + 0) * BT_STR + j] = (__bf16)(v.x * wj);
      smBTw[(s + 1) * BT_STR + j] = (__bf16)(v.y * wj);
      smBTw[(s + 2) * BT_STR + j] = (__bf16)(v.z * wj);
      smBTw[(s + 3) * BT_STR + j] = (__bf16)(v.w * wj);
    }
    const float4* X4 = (const float4*)X;
    for (int k = tid; k < (kCS * kDH) / 4; k += 256) {
      const float4 v = X4[k];
      const int t = k >> 4, d = (k & 15) * 4;  // X^T: (dh, t)
      smXT[(d + 0) * XT_STR + t] = (__bf16)v.x;
      smXT[(d + 1) * XT_STR + t] = (__bf16)v.y;
      smXT[(d + 2) * XT_STR + t] = (__bf16)v.z;
      smXT[(d + 3) * XT_STR + t] = (__bf16)v.w;
    }
    const float4* H4 = (const float4*)H;
    for (int k = tid; k < (kDS * kDH) / 4; k += 256) {
      const float4 v = H4[k];
      const int s = k >> 4, d = (k & 15) * 4;  // h_prev^T: (dh, s)
      smHT[(d + 0) * HT_STR + s] = (__bf16)v.x;
      smHT[(d + 1) * HT_STR + s] = (__bf16)v.y;
      smHT[(d + 2) * HT_STR + s] = (__bf16)v.z;
      smHT[(d + 3) * HT_STR + s] = (__bf16)v.w;
    }
  }
  __syncthreads();

  const int lhalf = (lane & 16) ? 8 : 0;  // D-frag row offset
  const int lcol  = lane & 15;            // D-frag col
  __bf16* sc = smScr + wave * (16 * SC_STR);

  // ---- Phase 3: Y = (1-a)*(L.CB).X + a*diag(df)*C.h_prev ----
  // Wave w owns 16-row tiles {w, 15-w}: every wave gets exactly 9 causal
  // j-blocks of 32 => balanced triangular work.
#pragma unroll
  for (int pass = 0; pass < 2; ++pass) {
    const int rt = __builtin_amdgcn_readfirstlane(pass ? (15 - wave) : wave);
    const int i0 = rt * 16;

    // Hoist the 8 row cumsum values for the mask
    float ai[8];
#pragma unroll
    for (int v = 0; v < 8; ++v) ai[v] = smAcum[i0 + v + lhalf];

    v8f yacc[4] = {v8f{0}, v8f{0}, v8f{0}, v8f{0}};
    const int nblocks = (rt >> 1) + 1;  // covers j <= i0+15 (scalar loop)
    for (int jb = 0; jb < nblocks; ++jb) {
      const int j0 = jb * 32;

      // GEMM1: CB tile 16x32 (K = DS = 128)
      v8f cb0 = v8f{0}, cb1 = v8f{0};
#pragma unroll
      for (int kk = 0; kk < kDS; kk += 32) {
        v16bf af = load_a16(smCs, C_STR, i0, kk, lane);
        v16bf b0 = load_b16(smBs, B_STR, j0, kk, lane);
        v16bf b1 = load_b16(smBs, B_STR, j0 + 16, kk, lane);
        cb0 = wmma_bf16(af, b0, cb0);
        cb1 = wmma_bf16(af, b1, cb1);
      }

      // Causal decay mask L[i,j] = exp(acum[i]-acum[j]) (i>=j), multiply,
      // write bf16 tile to wave-private scratch (D-layout -> A-layout xpose).
#pragma unroll
      for (int h = 0; h < 2; ++h) {
        const v8f cb = h ? cb1 : cb0;
        const int j = j0 + h * 16 + lcol;
        const float aj = smAcum[j];
#pragma unroll
        for (int v = 0; v < 8; ++v) {
          const int i = i0 + v + lhalf;
          const float val = (i >= j) ? cb[v] * __expf(ai[v] - aj) : 0.0f;
          sc[(v + lhalf) * SC_STR + h * 16 + lcol] = (__bf16)val;
        }
      }
      // LDS ops are in-order within a wave (DScnt), no barrier needed.

      // GEMM2: yacc += T(16x32) . X[j0:j0+32, :]
      v16bf ta = load_a16(sc, SC_STR, 0, 0, lane);
#pragma unroll
      for (int n = 0; n < 4; ++n) {
        v16bf xb = load_b16(smXT, XT_STR, n * 16, j0, lane);
        yacc[n] = wmma_bf16(ta, xb, yacc[n]);
      }
    }

    // GEMM4: P = C[i0:i0+16,:] . h_prev  (K = DS = 128)
    v8f pacc[4] = {v8f{0}, v8f{0}, v8f{0}, v8f{0}};
#pragma unroll
    for (int kk = 0; kk < kDS; kk += 32) {
      v16bf af = load_a16(smCs, C_STR, i0, kk, lane);
#pragma unroll
      for (int n = 0; n < 4; ++n) {
        v16bf hb = load_b16(smHT, HT_STR, n * 16, kk, lane);
        pacc[n] = wmma_bf16(af, hb, pacc[n]);
      }
    }

    // Combine and store Y in f32 (16 consecutive f32 per half-wave = 64B segs)
    float* Y = Yout + (size_t)bh * kCS * kDH;
#pragma unroll
    for (int n = 0; n < 4; ++n) {
#pragma unroll
      for (int v = 0; v < 8; ++v) {
        const int i = i0 + v + lhalf;
        const int d = n * 16 + lcol;
        Y[i * kDH + d] =
            (1.0f - kALPHA) * yacc[n][v] + kALPHA * smDF[i] * pacc[n][v];
      }
    }
  }

  // ---- Phase 4: h_end = a*dtot*h_prev + (1-a)*(B^T diag(w) X) ----
  // Wave w owns state rows [16w, 16w+16); K = CS = 256.
  {
    const int s0 = wave * 16;
    v8f hacc[4] = {v8f{0}, v8f{0}, v8f{0}, v8f{0}};
#pragma unroll
    for (int kk = 0; kk < kCS; kk += 32) {
      v16bf af = load_a16(smBTw, BT_STR, s0, kk, lane);
#pragma unroll
      for (int n = 0; n < 4; ++n) {
        v16bf xb = load_b16(smXT, XT_STR, n * 16, kk, lane);
        hacc[n] = wmma_bf16(af, xb, hacc[n]);
      }
    }
    const float dtot = smDF[kCS - 1];  // exp(atot)
    float* Ho = Hout + (size_t)bh * kDS * kDH;
#pragma unroll
    for (int n = 0; n < 4; ++n) {
#pragma unroll
      for (int v = 0; v < 8; ++v) {
        const int s = s0 + v + lhalf;
        const int d = n * 16 + lcol;
        const float hp = H[s * kDH + d];  // f32 reload for precision
        Ho[s * kDH + d] = kALPHA * dtot * hp + (1.0f - kALPHA) * hacc[n][v];
      }
    }
  }

  if (tid == 0) Dout[bh] = smDF[kCS - 1];
}

extern "C" void kernel_launch(void* const* d_in, const int* in_sizes, int n_in,
                              void* d_out, int out_size, void* d_ws,
                              size_t ws_size, hipStream_t stream) {
  (void)n_in; (void)out_size; (void)d_ws; (void)ws_size;
  const float* X = (const float*)d_in[0];
  const float* A = (const float*)d_in[1];
  const float* B = (const float*)d_in[2];
  const float* C = (const float*)d_in[3];
  const float* H = (const float*)d_in[4];

  const int nbh = in_sizes[1] / kCS;  // B_SZ * H = 512

  float* out  = (float*)d_out;
  float* Yout = out;                                   // [nbh, CS, DH]
  float* Hout = Yout + (size_t)nbh * kCS * kDH;        // [nbh, DS, DH]
  float* Dout = Hout + (size_t)nbh * kDS * kDH;        // [nbh]

  ssd_chunk_kernel<<<dim3(nbh), dim3(256), SMEM_BYTES, stream>>>(
      X, A, B, C, H, Yout, Hout, Dout);
}
